// SubgoalHistoryEncoder_85263690760481
// MI455X (gfx1250) — compile-verified, matrix operand order checked by reading, not run on hardware
//
#include <hip/hip_runtime.h>
#include <hip/hip_bf16.h>

#define T_SEQ   4096
#define T1      4097          // T + SOS
#define MP      4160          // padded rows (multiple of 64); attn keys reach 4127 < MP
#define DMODEL  512
#define NHEAD   8
#define HD      64
#define NQB     257           // ceil(T1/16)

typedef _Float16 v16h __attribute__((ext_vector_type(16)));
typedef _Float16 v8h  __attribute__((ext_vector_type(8)));
typedef float    v8f  __attribute__((ext_vector_type(8)));

enum { OUT_F32 = 0, OUT_F16 = 1, OUT_F16T = 2 };

// A/B fragment for v_wmma_f32_16x16x32_f16, row-major source:
// lane (0..31): row/col = lane&15 ; elements 0-7 = K kb..kb+7, 8-15 = K kb+16..kb+23
// where kb = (lane<16 ? 0 : 8). Two contiguous 16B loads.
__device__ __forceinline__ v16h load_frag(const _Float16* __restrict__ base, int kb) {
  v8h lo = *(const v8h*)(base + kb);
  v8h hi = *(const v8h*)(base + kb + 16);
  return __builtin_shufflevector(lo, hi, 0,1,2,3,4,5,6,7,8,9,10,11,12,13,14,15);
}

__device__ __forceinline__ v8f wmma32(v16h a, v16h b, v8f c) {
  return __builtin_amdgcn_wmma_f32_16x16x32_f16(false, a, false, b, (short)0, c, false, false);
}

// ---------------------------------------------------------------- prep: bid1 / pos
__global__ __launch_bounds__(256)
void prep_kernel(const int* __restrict__ batch_id, int* __restrict__ bid1, int* __restrict__ pos) {
  int t = blockIdx.x * 256 + threadIdx.x;
  if (t >= MP) return;
  if (t == 0) { bid1[0] = -1; pos[0] = 0; }
  else if (t < T1) {
    int b = batch_id[t - 1];
    bid1[t] = b;
    int lo = 0, hi = T_SEQ;                      // lower_bound (searchsorted left)
    while (lo < hi) { int mid = (lo + hi) >> 1; if (batch_id[mid] < b) lo = mid + 1; else hi = mid; }
    pos[t] = (t - 1) - lo;
  } else { bid1[t] = -3; }
}

// ---------------------------------------------------------------- embeddings + PE
__global__ __launch_bounds__(256)
void embed_kernel(const int* __restrict__ aseq, const float* __restrict__ Wt,
                  const float* __restrict__ bt, const float* __restrict__ Wa,
                  const float* __restrict__ ba, const float* __restrict__ sos,
                  const int* __restrict__ pos,
                  float* __restrict__ x32, _Float16* __restrict__ x16) {
  int idx = blockIdx.x * 256 + threadIdx.x;
  if (idx >= MP * DMODEL) return;
  int t = idx >> 9, d = idx & 511;
  float v;
  if (t == 0) v = sos[d];
  else if (t < T1) {
    int i = t - 1;
    int a0 = aseq[i * 2 + 0];
    int a1 = aseq[i * 2 + 1] + 1;
    float ang = (float)pos[t] * __expf(-9.2103403719761836f * (float)(d & ~1) / 512.0f);
    float pe  = (d & 1) ? __cosf(ang) : __sinf(ang);
    v = Wt[a0 * DMODEL + d] + bt[d] + Wa[a1 * DMODEL + d] + ba[d] + pe;
  } else v = 0.0f;                               // zero pad rows for WMMA tiles
  x32[idx] = v;
  x16[idx] = (_Float16)v;
}

// ---------------------------------------------------------------- transpose-convert all 12 W -> f16 (batched)
__global__ __launch_bounds__(256)
void wconv_kernel(const float* __restrict__ Wq, const float* __restrict__ Wk,
                  const float* __restrict__ Wv, const float* __restrict__ Wo,
                  const float* __restrict__ W1, const float* __restrict__ W2,
                  _Float16* __restrict__ wT) {
  int idx = blockIdx.x * 256 + threadIdx.x;      // 512*512 per matrix
  int mat = blockIdx.y;                          // 0..11 = l*6 + j
  int l = mat / 6, j = mat % 6;
  const float* srcs[6] = { Wq, Wk, Wv, Wo, W1, W2 };
  const float* W = srcs[j] + (size_t)l * DMODEL * DMODEL;
  int n = idx >> 9, k = idx & 511;
  wT[(size_t)mat * DMODEL * DMODEL + idx] = (_Float16)W[k * DMODEL + n];  // Wt[n][k] = W[k][n]
}

// ---------------------------------------------------------------- WMMA GEMM  D = A(MPx512) @ W(512x512) + bias
// A f16 row-major, Wt f16 transposed [n][k]. One wave -> 64x32 tile (8 wmma : 12 loads per kstep).
__global__ __launch_bounds__(256)
void gemm_kernel(const _Float16* __restrict__ A, const _Float16* __restrict__ Wt,
                 const float* __restrict__ bias, void* __restrict__ out,
                 int mode, int relu) {
  int lane = threadIdx.x & 31;
  int gw   = blockIdx.x * 8 + (threadIdx.x >> 5);
  const int NTILES = (MP / 64) * (DMODEL / 32);  // 65*16 = 1040
  if (gw >= NTILES) return;
  int mt = gw >> 4, nt = gw & 15;
  int mbase = mt * 64, nbase = nt * 32;
  int lr = lane & 15;
  int kb = (lane < 16) ? 0 : 8;

  const _Float16* arow[4];
  #pragma unroll
  for (int i = 0; i < 4; ++i) arow[i] = A + (size_t)(mbase + i * 16 + lr) * DMODEL;
  const _Float16* brow0 = Wt + (size_t)(nbase +      lr) * DMODEL;
  const _Float16* brow1 = Wt + (size_t)(nbase + 16 + lr) * DMODEL;

  v8f c[4][2] = {};
  #pragma unroll 2
  for (int kk = 0; kk < DMODEL; kk += 32) {
    v16h b0 = load_frag(brow0, kk + kb);
    v16h b1 = load_frag(brow1, kk + kb);
    #pragma unroll
    for (int i = 0; i < 4; ++i) {
      v16h a = load_frag(arow[i], kk + kb);
      c[i][0] = wmma32(a, b0, c[i][0]);
      c[i][1] = wmma32(a, b1, c[i][1]);
    }
  }

  int   mhalf = (lane < 16) ? 0 : 8;
  float bs0 = bias[nbase + lr], bs1 = bias[nbase + 16 + lr];
  int   n0 = nbase + lr, n1 = nbase + 16 + lr;
  #pragma unroll
  for (int i = 0; i < 4; ++i) {
    #pragma unroll
    for (int r = 0; r < 8; ++r) {
      int m = mbase + i * 16 + mhalf + r;
      float v0 = c[i][0][r] + bs0;
      float v1 = c[i][1][r] + bs1;
      if (relu) { v0 = fmaxf(v0, 0.f); v1 = fmaxf(v1, 0.f); }
      if (mode == OUT_F32) {
        float* o = (float*)out;
        o[(size_t)m * DMODEL + n0] = v0;
        o[(size_t)m * DMODEL + n1] = v1;
      } else if (mode == OUT_F16) {
        _Float16* o = (_Float16*)out;
        bool z = (m >= T1);                      // keep pad rows zero
        o[(size_t)m * DMODEL + n0] = z ? (_Float16)0.f : (_Float16)v0;
        o[(size_t)m * DMODEL + n1] = z ? (_Float16)0.f : (_Float16)v1;
      } else {                                   // OUT_F16T : out[n][m], [512][MP]
        _Float16* o = (_Float16*)out;
        bool z = (m >= T1);
        o[(size_t)n0 * MP + m] = z ? (_Float16)0.f : (_Float16)v0;
        o[(size_t)n1 * MP + m] = z ? (_Float16)0.f : (_Float16)v1;
      }
    }
  }
}

// ---------------------------------------------------------------- flash attention
// one wave = one (head, 16-query block); K streamed in 32-key blocks; online softmax.
__global__ __launch_bounds__(256)
void attn_kernel(const _Float16* __restrict__ Q, const _Float16* __restrict__ K,
                 const _Float16* __restrict__ Vt, const int* __restrict__ bid1,
                 _Float16* __restrict__ Oa) {
  __shared__ _Float16 pt_all[8][16 * 32];        // per-wave P tile (C-layout -> A-layout via LDS)
  int wid  = threadIdx.x >> 5;
  int lane = threadIdx.x & 31;
  int gw   = blockIdx.x * 8 + wid;
  if (gw >= NHEAD * NQB) return;
  int h = gw / NQB, qb = gw % NQB;
  int qbase = qb * 16;
  int lr = lane & 15;
  int kb = (lane < 16) ? 0 : 8;
  int mhalf = (lane < 16) ? 0 : 8;
  _Float16* pt = pt_all[wid];

  int qrow = qbase + lr; if (qrow >= T1) qrow = T1 - 1;   // clamp; masked rows never stored
  const _Float16* qp = Q + (size_t)qrow * DMODEL + h * HD;
  v16h aq0 = load_frag(qp, kb);
  v16h aq1 = load_frag(qp, 32 + kb);

  int   bidq[8];
  float mrow[8], lrow[8];
  #pragma unroll
  for (int r = 0; r < 8; ++r) {
    int q = qbase + mhalf + r;
    bidq[r] = (q < T1) ? bid1[q] : -4;
    mrow[r] = -__builtin_inff();
    lrow[r] = 0.f;
  }
  v8f oacc[4] = {};

  int qlast = qbase + 15; if (qlast > T1 - 1) qlast = T1 - 1;
  const float scale = 0.125f;                    // 1/sqrt(64)

  for (int kb2 = 0; kb2 <= qlast; kb2 += 32) {
    // S = Q K^T (two 16x16 subtiles over 32 keys)
    const _Float16* kp0 = K + (size_t)(kb2 +      lr) * DMODEL + h * HD;
    const _Float16* kp1 = K + (size_t)(kb2 + 16 + lr) * DMODEL + h * HD;
    v8f s0 = {}, s1 = {};
    s0 = wmma32(aq0, load_frag(kp0, kb),      s0);
    s0 = wmma32(aq1, load_frag(kp0, 32 + kb), s0);
    s1 = wmma32(aq0, load_frag(kp1, kb),      s1);
    s1 = wmma32(aq1, load_frag(kp1, 32 + kb), s1);

    int key0 = kb2 + lr, key1 = kb2 + 16 + lr;   // both < MP always
    int bidk0 = bid1[key0], bidk1 = bid1[key1];

    float sm0[8], sm1[8], nm[8], rs[8];
    #pragma unroll
    for (int r = 0; r < 8; ++r) {
      int q = qbase + mhalf + r;
      float x0 = s0[r] * scale;
      float x1 = s1[r] * scale;
      bool a0 = (key0 == 0) || ((q >= key0) && (bidq[r] == bidk0));
      bool a1 = (key1 == 0) || ((q >= key1) && (bidq[r] == bidk1));
      x0 = a0 ? x0 : -1e9f;
      x1 = a1 ? x1 : -1e9f;
      sm0[r] = x0; sm1[r] = x1;
      float mx = fmaxf(x0, x1);
      #pragma unroll
      for (int d = 1; d < 16; d <<= 1) mx = fmaxf(mx, __shfl_xor(mx, d, 32));
      nm[r] = fmaxf(mrow[r], mx);
    }
    #pragma unroll
    for (int r = 0; r < 8; ++r) {
      float p0 = __expf(sm0[r] - nm[r]);
      float p1 = __expf(sm1[r] - nm[r]);
      int row = mhalf + r;
      pt[row * 32 +      lr] = (_Float16)p0;
      pt[row * 32 + 16 + lr] = (_Float16)p1;
      float ssum = p0 + p1;
      #pragma unroll
      for (int d = 1; d < 16; d <<= 1) ssum += __shfl_xor(ssum, d, 32);
      rs[r] = ssum;
    }
    #pragma unroll
    for (int r = 0; r < 8; ++r) {
      float a = __expf(mrow[r] - nm[r]);
      lrow[r] = lrow[r] * a + rs[r];
      mrow[r] = nm[r];
      oacc[0][r] *= a; oacc[1][r] *= a; oacc[2][r] *= a; oacc[3][r] *= a;
    }
    // wave-local LDS RAW fence: P stores must land before A-layout reload
    asm volatile("s_wait_dscnt 0" ::: "memory");
    v16h ap = load_frag(pt + lr * 32, kb);       // P as A-fragment (16x32, K=keys)
    #pragma unroll
    for (int j = 0; j < 4; ++j) {                // O += P @ V  (V transposed: Vt[d][t])
      const _Float16* vp = Vt + (size_t)(h * HD + j * 16 + lr) * MP + kb2;
      oacc[j] = wmma32(ap, load_frag(vp, kb), oacc[j]);
    }
  }

  #pragma unroll
  for (int r = 0; r < 8; ++r) {
    int q = qbase + mhalf + r;
    if (q < T1) {
      float inv = 1.0f / lrow[r];
      _Float16* op = Oa + (size_t)q * DMODEL + h * HD + lr;
      op[0]  = (_Float16)(oacc[0][r] * inv);
      op[16] = (_Float16)(oacc[1][r] * inv);
      op[32] = (_Float16)(oacc[2][r] * inv);
      op[48] = (_Float16)(oacc[3][r] * inv);
    }
  }
}

// ---------------------------------------------------------------- residual + layernorm (one row per block)
__global__ __launch_bounds__(256)
void resid_ln_kernel(float* __restrict__ x32, _Float16* __restrict__ x16,
                     const float* __restrict__ res, const float* __restrict__ g,
                     const float* __restrict__ b) {
  int row = blockIdx.x;                          // < T1
  int tid = threadIdx.x;
  __shared__ float red[2][8];
  size_t base = (size_t)row * DMODEL;
  float v0 = x32[base + tid]       + res[base + tid];
  float v1 = x32[base + 256 + tid] + res[base + 256 + tid];
  float s = v0 + v1, sq = v0 * v0 + v1 * v1;
  #pragma unroll
  for (int d = 1; d < 32; d <<= 1) { s += __shfl_xor(s, d, 32); sq += __shfl_xor(sq, d, 32); }
  int wid = tid >> 5, lane = tid & 31;
  if (lane == 0) { red[0][wid] = s; red[1][wid] = sq; }
  __syncthreads();
  if (wid == 0) {
    float a = (lane < 8) ? red[0][lane] : 0.f;
    float c = (lane < 8) ? red[1][lane] : 0.f;
    #pragma unroll
    for (int d = 1; d < 8; d <<= 1) { a += __shfl_xor(a, d, 32); c += __shfl_xor(c, d, 32); }
    if (lane == 0) { red[0][0] = a; red[1][0] = c; }
  }
  __syncthreads();
  float mean = red[0][0] * (1.0f / DMODEL);
  float var  = red[1][0] * (1.0f / DMODEL) - mean * mean;
  float rstd = rsqrtf(var + 1e-5f);
  float y0 = (v0 - mean) * rstd * g[tid]       + b[tid];
  float y1 = (v1 - mean) * rstd * g[256 + tid] + b[256 + tid];
  x32[base + tid] = y0;          x32[base + 256 + tid] = y1;
  x16[base + tid] = (_Float16)y0; x16[base + 256 + tid] = (_Float16)y1;
}

// ---------------------------------------------------------------- final copy
__global__ __launch_bounds__(256)
void copyout_kernel(const float* __restrict__ x32, float* __restrict__ out) {
  int idx = blockIdx.x * 256 + threadIdx.x;
  if (idx < T1 * DMODEL) out[idx] = x32[idx];
}

// ================================================================ host
extern "C" void kernel_launch(void* const* d_in, const int* in_sizes, int n_in,
                              void* d_out, int out_size, void* d_ws, size_t ws_size,
                              hipStream_t stream) {
  const int*   aseq     = (const int*)  d_in[0];
  const int*   batch_id = (const int*)  d_in[1];
  const float* W_type   = (const float*)d_in[2];
  const float* b_type   = (const float*)d_in[3];
  const float* W_arg    = (const float*)d_in[4];
  const float* b_arg    = (const float*)d_in[5];
  const float* sos      = (const float*)d_in[6];
  const float* Wq = (const float*)d_in[7];  const float* bq = (const float*)d_in[8];
  const float* Wk = (const float*)d_in[9];  const float* bk = (const float*)d_in[10];
  const float* Wv = (const float*)d_in[11]; const float* bv = (const float*)d_in[12];
  const float* Wo = (const float*)d_in[13]; const float* bo = (const float*)d_in[14];
  const float* ln1_g = (const float*)d_in[15]; const float* ln1_b = (const float*)d_in[16];
  const float* W1 = (const float*)d_in[17]; const float* b1 = (const float*)d_in[18];
  const float* W2 = (const float*)d_in[19]; const float* b2 = (const float*)d_in[20];
  const float* ln2_g = (const float*)d_in[21]; const float* ln2_b = (const float*)d_in[22];

  char* ws = (char*)d_ws;
  size_t off = 0;
  auto alloc = [&](size_t bytes) { size_t o = off; off += (bytes + 255) & ~(size_t)255; return o; };

  int*      bid1   = (int*)      (ws + alloc((size_t)MP * 4));
  int*      pos    = (int*)      (ws + alloc((size_t)MP * 4));
  float*    x32    = (float*)    (ws + alloc((size_t)MP * DMODEL * 4));
  _Float16* x16    = (_Float16*) (ws + alloc((size_t)MP * DMODEL * 2));
  _Float16* q16    = (_Float16*) (ws + alloc((size_t)MP * DMODEL * 2));
  _Float16* k16    = (_Float16*) (ws + alloc((size_t)MP * DMODEL * 2));
  _Float16* v16t   = (_Float16*) (ws + alloc((size_t)DMODEL * MP * 2));
  _Float16* attn16 = (_Float16*) (ws + alloc((size_t)MP * DMODEL * 2));
  float*    res32  = (float*)    (ws + alloc((size_t)MP * DMODEL * 4));
  _Float16* h16    = (_Float16*) (ws + alloc((size_t)MP * DMODEL * 2));
  _Float16* wT     = (_Float16*) (ws + alloc((size_t)12 * DMODEL * DMODEL * 2));

  // pad rows of attn16 stay zero (attention only writes q < T1)
  hipMemsetAsync(attn16, 0, (size_t)MP * DMODEL * 2, stream);

  prep_kernel<<<(MP + 255) / 256, 256, 0, stream>>>(batch_id, bid1, pos);
  embed_kernel<<<(MP * DMODEL + 255) / 256, 256, 0, stream>>>(
      aseq, W_type, b_type, W_arg, b_arg, sos, pos, x32, x16);

  dim3 wcg((DMODEL * DMODEL) / 256, 12);
  wconv_kernel<<<wcg, 256, 0, stream>>>(Wq, Wk, Wv, Wo, W1, W2, wT);

  const int GEMM_BLOCKS = (MP / 64) * (DMODEL / 32) / 8;   // 130
  for (int l = 0; l < 2; ++l) {
    _Float16* wTq = wT + (size_t)(l * 6 + 0) * DMODEL * DMODEL;
    _Float16* wTk = wT + (size_t)(l * 6 + 1) * DMODEL * DMODEL;
    _Float16* wTv = wT + (size_t)(l * 6 + 2) * DMODEL * DMODEL;
    _Float16* wTo = wT + (size_t)(l * 6 + 3) * DMODEL * DMODEL;
    _Float16* wT1 = wT + (size_t)(l * 6 + 4) * DMODEL * DMODEL;
    _Float16* wT2 = wT + (size_t)(l * 6 + 5) * DMODEL * DMODEL;

    gemm_kernel<<<GEMM_BLOCKS, 256, 0, stream>>>(x16, wTq, bq + l * DMODEL, q16,  OUT_F16,  0);
    gemm_kernel<<<GEMM_BLOCKS, 256, 0, stream>>>(x16, wTk, bk + l * DMODEL, k16,  OUT_F16,  0);
    gemm_kernel<<<GEMM_BLOCKS, 256, 0, stream>>>(x16, wTv, bv + l * DMODEL, v16t, OUT_F16T, 0);

    attn_kernel<<<NQB, 256, 0, stream>>>(q16, k16, v16t, bid1, attn16);

    gemm_kernel<<<GEMM_BLOCKS, 256, 0, stream>>>(attn16, wTo, bo + l * DMODEL, res32, OUT_F32, 0);
    resid_ln_kernel<<<T1, 256, 0, stream>>>(x32, x16, res32, ln1_g + l * DMODEL, ln1_b + l * DMODEL);

    gemm_kernel<<<GEMM_BLOCKS, 256, 0, stream>>>(x16, wT1, b1 + l * DMODEL, h16,   OUT_F16, 1);
    gemm_kernel<<<GEMM_BLOCKS, 256, 0, stream>>>(h16, wT2, b2 + l * DMODEL, res32, OUT_F32, 0);
    resid_ln_kernel<<<T1, 256, 0, stream>>>(x32, x16, res32, ln2_g + l * DMODEL, ln2_b + l * DMODEL);
  }

  copyout_kernel<<<(T1 * DMODEL + 255) / 256, 256, 0, stream>>>(x32, (float*)d_out);
}
